// GRUDec_25142738551253
// MI455X (gfx1250) — matrix operand order repaired and checked by reading, not compiled, not run on hardware
//
#include <hip/hip_runtime.h>
#include <hip/hip_bf16.h>

#define BB 32
#define LL 64
#define DD 512
#define GG 1536   // 3*D
#define VV 50257
#define LOUT 63   // L-1

typedef __attribute__((ext_vector_type(16))) _Float16 v16h;
typedef __attribute__((ext_vector_type(8)))  _Float16 v8h;
typedef __attribute__((ext_vector_type(8)))  float    v8f;

union V16H { v16h v; v8h half8[2]; _Float16 e[16]; };

// ---------------------------------------------------------------------------
// WMMA tile accumulator:  C(16x16,f32) += A_tile(16x32,f16) x B_tile(32x16,f16)
// over K=512.  A is PRE-CONVERTED f16, row-major [M x 512].  W is f32
// row-major [N x 512], so B(k,n) = W[n*512 + k], converted on the fly.
// Layouts used here (contiguous k order, same for A and B):
//   A: lane m = lane&15;  e[0..7]  = A[m][kk + 16*laneHi + 0..7]   (16B run)
//                         e[8..15] = A[m][kk + 16*laneHi + 8 + ..] (16B run)
//   B: lane n = lane&15;  e[h]     = W[n][kk + 16*laneHi + h]
//   C: vgpr i -> row i + 8*laneHi, col = lane&15
// Out-of-range columns are CLAMPED to a valid row (branchless); garbage flows
// only into output columns the caller's store guard discards.
// ---------------------------------------------------------------------------
__device__ inline v8f gemm_tile_f16(const _Float16* __restrict__ A,
                                    const float* __restrict__ W,
                                    int m0, int n0, int Nmax, v8f c) {
    const int lane   = threadIdx.x & 31;
    const int laneHi = lane >> 4;
    const int lanLo  = lane & 15;
    const int m      = m0 + lanLo;
    const int n      = n0 + lanLo;
    const int nc     = (n < Nmax) ? n : (Nmax - 1);   // clamp, no branch
    const _Float16* __restrict__ Arow = A + (size_t)m  * DD + laneHi * 16;
    const float*    __restrict__ Wrow = W + (size_t)nc * DD + laneHi * 16;

#pragma unroll 4
    for (int kk = 0; kk < DD; kk += 32) {
        // Prefetch next K-chunk of the streaming W operand.
        __builtin_prefetch(Wrow + kk + 64, 0, 3);

        V16H a, b;
        // A tile: two contiguous 8-half (16B) runs -> 2 x b128, no cvt
        a.half8[0] = *(const v8h*)(Arow + kk);
        a.half8[1] = *(const v8h*)(Arow + kk + 8);
        // B tile: 16 contiguous floats per lane -> 4 x b128 + cvt
#pragma unroll
        for (int q = 0; q < 4; ++q) {
            const float4 f4 = *(const float4*)(Wrow + kk + 4 * q);
            b.e[4 * q + 0] = (_Float16)f4.x;
            b.e[4 * q + 1] = (_Float16)f4.y;
            b.e[4 * q + 2] = (_Float16)f4.z;
            b.e[4 * q + 3] = (_Float16)f4.w;
        }
        c = __builtin_amdgcn_wmma_f32_16x16x32_f16(
                /*neg_a=*/false, a.v, /*neg_b=*/false, b.v,
                /*c_mod=*/(short)0, c, /*reuse_a=*/false, /*reuse_b=*/false);
    }
    return c;
}

// ---------------------------------------------------------------------------
// Kernel 1: x0 = relu(emb[ids[:,0]]) (f16) ;
//           h0 = hidden . Wb + bb   (f32 for GRU + f16 for WMMA)   [32 x 512]
// ---------------------------------------------------------------------------
__global__ void k_prep(const int* __restrict__ ids,
                       const float* __restrict__ hidden,
                       const float* __restrict__ emb,
                       const float* __restrict__ Wb,
                       const float* __restrict__ bb,
                       _Float16* __restrict__ x0h,
                       float* __restrict__ h0,
                       _Float16* __restrict__ h0h) {
    __shared__ float sWb[LL];
    const int b = blockIdx.x;
    const int t = threadIdx.x;
    if (t < LL) sWb[t] = Wb[t];
    __syncthreads();
    const int tok = ids[b * LL];
    const float bias = bb[0];
    for (int d = t; d < DD; d += 256) {
        float xv = emb[(size_t)tok * DD + d];
        x0h[b * DD + d] = (_Float16)(xv > 0.f ? xv : 0.f);
        float acc = bias;
        const float* hb = hidden + ((size_t)b * LL) * DD + d;
#pragma unroll 8
        for (int l = 0; l < LL; ++l) acc += hb[(size_t)l * DD] * sWb[l];
        h0[b * DD + d]  = acc;
        h0h[b * DD + d] = (_Float16)acc;
    }
}

// ---------------------------------------------------------------------------
// Kernel 2: gx = x0 @ W_ih^T + b_ih ; gh = h0 @ W_hh^T + b_hh   [32 x 1536]
// grid = (96 n-tiles, 2 which), block = 64 (2 waves = 2 M-tiles)
// ---------------------------------------------------------------------------
__global__ void k_gates(const _Float16* __restrict__ x0h, const _Float16* __restrict__ h0h,
                        const float* __restrict__ W_ih, const float* __restrict__ W_hh,
                        const float* __restrict__ b_ih, const float* __restrict__ b_hh,
                        float* __restrict__ gx, float* __restrict__ gh) {
    const _Float16* A    = (blockIdx.y == 0) ? x0h  : h0h;
    const float*    W    = (blockIdx.y == 0) ? W_ih : W_hh;
    const float*    bias = (blockIdx.y == 0) ? b_ih : b_hh;
    float*          G    = (blockIdx.y == 0) ? gx   : gh;

    const int n0 = blockIdx.x * 16;
    const int m0 = (threadIdx.x >> 5) * 16;
    v8f c = {};
    c = gemm_tile_f16(A, W, m0, n0, GG, c);

    const int lane   = threadIdx.x & 31;
    const int laneHi = lane >> 4;
    const int col    = n0 + (lane & 15);
    const float bv   = bias[col];
#pragma unroll
    for (int i = 0; i < 8; ++i) {
        const int row = m0 + i + laneHi * 8;
        G[(size_t)row * GG + col] = c[i] + bv;
    }
}

// ---------------------------------------------------------------------------
// Kernel 3: GRU cell elementwise -> h1 (f16, only consumer is WMMA) [32 x 512]
// ---------------------------------------------------------------------------
__global__ void k_gru(const float* __restrict__ gx, const float* __restrict__ gh,
                      const float* __restrict__ h0, _Float16* __restrict__ h1h) {
    const int idx = blockIdx.x * 256 + threadIdx.x;   // 16384 total
    const int b = idx >> 9;
    const int d = idx & 511;
    const float gxr = gx[(size_t)b * GG + d];
    const float gxz = gx[(size_t)b * GG + DD + d];
    const float gxn = gx[(size_t)b * GG + 2 * DD + d];
    const float ghr = gh[(size_t)b * GG + d];
    const float ghz = gh[(size_t)b * GG + DD + d];
    const float ghn = gh[(size_t)b * GG + 2 * DD + d];
    const float r = 1.f / (1.f + __expf(-(gxr + ghr)));
    const float z = 1.f / (1.f + __expf(-(gxz + ghz)));
    const float n = tanhf(gxn + r * ghn);
    h1h[idx] = (_Float16)((1.f - z) * n + z * h0[idx]);
}

// ---------------------------------------------------------------------------
// Kernel 4: logits = h1 @ Wp^T + bp, fused 63-way broadcast store (NT b32;
// V is odd so vector stores would be misaligned).  Lanes 0-15 cover 16
// consecutive columns -> 64B coalesced segments per half-wave.
// grid = 3142 n-tiles, block = 64 (2 waves = 2 M-tiles sharing Wp lines)
// ---------------------------------------------------------------------------
__global__ void k_proj(const _Float16* __restrict__ h1h, const float* __restrict__ Wp,
                       const float* __restrict__ bp, float* __restrict__ out) {
    const int n0 = blockIdx.x * 16;
    const int m0 = (threadIdx.x >> 5) * 16;
    v8f c = {};
    c = gemm_tile_f16(h1h, Wp, m0, n0, VV, c);

    const int lane   = threadIdx.x & 31;
    const int laneHi = lane >> 4;
    const int col    = n0 + (lane & 15);
    if (col < VV) {
        const float bv = bp[col];
#pragma unroll
        for (int i = 0; i < 8; ++i) {
            const float val = c[i] + bv;
            const size_t row = (size_t)(m0 + i + laneHi * 8);   // batch index
            float* op = out + row * (size_t)LOUT * VV + col;
            for (int l = 0; l < LOUT; ++l) {
                __builtin_nontemporal_store(val, op);
                op += VV;
            }
        }
    }
}

// ---------------------------------------------------------------------------
extern "C" void kernel_launch(void* const* d_in, const int* in_sizes, int n_in,
                              void* d_out, int out_size, void* d_ws, size_t ws_size,
                              hipStream_t stream) {
    (void)in_sizes; (void)n_in; (void)out_size; (void)ws_size;

    const int*   ids    = (const int*)  d_in[0];
    const float* hidden = (const float*)d_in[1];
    const float* emb    = (const float*)d_in[2];
    const float* Wb     = (const float*)d_in[3];
    const float* bb     = (const float*)d_in[4];
    const float* W_ih   = (const float*)d_in[5];
    const float* b_ih   = (const float*)d_in[6];
    const float* W_hh   = (const float*)d_in[7];
    const float* b_hh   = (const float*)d_in[8];
    const float* Wp     = (const float*)d_in[9];
    const float* bp     = (const float*)d_in[10];
    float* out = (float*)d_out;

    float* ws = (float*)d_ws;
    float*     h0  = ws;                          // 32*512 f32
    float*     gx  = ws + 16384;                  // 32*1536 f32
    float*     gh  = ws + 65536;                  // 32*1536 f32
    _Float16*  x0h = (_Float16*)(ws + 114688);    // 32*512 f16
    _Float16*  h0h = (_Float16*)(ws + 122880);    // 32*512 f16
    _Float16*  h1h = (_Float16*)(ws + 131072);    // 32*512 f16

    k_prep <<<BB, 256, 0, stream>>>(ids, hidden, emb, Wb, bb, x0h, h0, h0h);
    k_gates<<<dim3(GG / 16, 2), 64, 0, stream>>>(x0h, h0h, W_ih, W_hh, b_ih, b_hh, gx, gh);
    k_gru  <<<(BB * DD) / 256, 256, 0, stream>>>(gx, gh, h0, h1h);
    k_proj <<<(VV + 15) / 16, 64, 0, stream>>>(h1h, Wp, bp, out);
}
